// Model_29781303231158
// MI455X (gfx1250) — compile-verified, matrix operand order checked by reading, not run on hardware
//
#include <hip/hip_runtime.h>

#define KSTEPS 10
#define ALPHA 0.1f
#define DIMS 128
#define TPW 8   // row-tiles per wave in the GEMM

typedef __attribute__((ext_vector_type(16))) __bf16 v16bf;
typedef __attribute__((ext_vector_type(8)))  float  v8f;

// ---------------- WMMA fragment helpers (ISA 7.12.2 layouts, wave32) ----------------

// A (16x32 bf16): lane = M (l&15); elems 0..7 -> K = khalf..khalf+7, elems 8..15 -> K+16,
// khalf = 8 for lanes 16..31.  Source row-major fp32 ld=128; vectorized float4 loads.
__device__ __forceinline__ v16bf load_a_frag_global(const float* __restrict__ rowbase,
                                                    int lane, int kt) {
    const int m     = lane & 15;
    const int khalf = (lane & 16) ? 8 : 0;
    const float* p  = rowbase + (size_t)m * DIMS + kt * 32 + khalf;
    float4 f0 = ((const float4*)p)[0];
    float4 f1 = ((const float4*)p)[1];
    float4 f2 = ((const float4*)(p + 16))[0];
    float4 f3 = ((const float4*)(p + 16))[1];
    v16bf a;
    a[0]  = (__bf16)f0.x; a[1]  = (__bf16)f0.y; a[2]  = (__bf16)f0.z; a[3]  = (__bf16)f0.w;
    a[4]  = (__bf16)f1.x; a[5]  = (__bf16)f1.y; a[6]  = (__bf16)f1.z; a[7]  = (__bf16)f1.w;
    a[8]  = (__bf16)f2.x; a[9]  = (__bf16)f2.y; a[10] = (__bf16)f2.z; a[11] = (__bf16)f2.w;
    a[12] = (__bf16)f3.x; a[13] = (__bf16)f3.y; a[14] = (__bf16)f3.z; a[15] = (__bf16)f3.w;
    return a;
}

// B fragments pre-swizzled in LDS: Wfrag[((kt*8+nt)*32 + lane)*16 + e] -> one 32B read/lane.
__device__ __forceinline__ v16bf load_b_frag_lds(const __bf16* __restrict__ Wfrag,
                                                 int lane, int kt, int nt) {
    return *(const v16bf*)(Wfrag + (size_t)(((kt * 8 + nt) * 32 + lane) * 16));
}

// ---------------- GEMM: Y[rows x 128] = act(X[rows x 128] @ W[128 x 128] + bias) --------
// 256 threads = 8 waves; each wave owns TPW consecutive 16-row tiles.  For each tile the
// whole 16x128 A block is register-cached before any store -> X may alias Y (in-place GEMM2).
template <bool RELU>
__global__ __launch_bounds__(256) void gemm_bias_kernel(const float* __restrict__ X,
                                                        const float* __restrict__ W,
                                                        const float* __restrict__ bias,
                                                        float* __restrict__ Y, int nrows) {
    __shared__ __bf16 Wfrag[4 * 8 * 32 * 16];  // 32 KB, fragment-swizzled
    __shared__ float  blds[DIMS];
    const int t = threadIdx.x;

    // Stage W (fp32 -> bf16) directly into WMMA-fragment layout (W is L2-hot, 64 KB).
    for (int idx = t; idx < 4 * 8 * 32 * 16; idx += 256) {
        int e    = idx & 15;
        int lane = (idx >> 4) & 31;
        int frag = idx >> 9;
        int nt   = frag & 7;
        int kt   = frag >> 3;
        int n     = nt * 16 + (lane & 15);
        int khalf = (lane & 16) ? 8 : 0;
        int k     = kt * 32 + khalf + ((e < 8) ? e : e + 8);
        Wfrag[idx] = (__bf16)W[(size_t)k * DIMS + n];
    }
    if (t < DIMS) blds[t] = bias[t];
    __syncthreads();

    const int lane = t & 31;
    const int wave = t >> 5;
    const int col_lane = lane & 15;
    const int row_off  = (lane & 16) ? 8 : 0;
    const int tile0 = (blockIdx.x * 8 + wave) * TPW;

    for (int i = 0; i < TPW; ++i) {
        const int row0 = (tile0 + i) * 16;
        if (row0 >= nrows) break;  // wave-uniform
        const float* rowbase = X + (size_t)row0 * DIMS;
        // Register-cache the entire 16x128 A block for this tile.
        v16bf a0 = load_a_frag_global(rowbase, lane, 0);
        v16bf a1 = load_a_frag_global(rowbase, lane, 1);
        v16bf a2 = load_a_frag_global(rowbase, lane, 2);
        v16bf a3 = load_a_frag_global(rowbase, lane, 3);
        // Single 64-bit base per tile; all store offsets become compile-time immediates.
        float* yb = Y + (size_t)(row0 + row_off) * DIMS + col_lane;
#pragma unroll
        for (int nt = 0; nt < 8; ++nt) {
            // Load all 4 B fragments first so DS loads clause and overlap the WMMAs.
            v16bf b0 = load_b_frag_lds(Wfrag, lane, 0, nt);
            v16bf b1 = load_b_frag_lds(Wfrag, lane, 1, nt);
            v16bf b2 = load_b_frag_lds(Wfrag, lane, 2, nt);
            v16bf b3 = load_b_frag_lds(Wfrag, lane, 3, nt);
            v8f acc = {};
            acc = __builtin_amdgcn_wmma_f32_16x16x32_bf16(false, a0, false, b0,
                                                          (short)0, acc, false, false);
            acc = __builtin_amdgcn_wmma_f32_16x16x32_bf16(false, a1, false, b1,
                                                          (short)0, acc, false, false);
            acc = __builtin_amdgcn_wmma_f32_16x16x32_bf16(false, a2, false, b2,
                                                          (short)0, acc, false, false);
            acc = __builtin_amdgcn_wmma_f32_16x16x32_bf16(false, a3, false, b3,
                                                          (short)0, acc, false, false);
            const float bval = blds[nt * 16 + col_lane];
#pragma unroll
            for (int r = 0; r < 8; ++r) {
                float v = acc[r] + bval;
                if (RELU) v = fmaxf(v, 0.0f);
                yb[r * DIMS + nt * 16] = v;   // immediate-offset stores off one base
            }
        }
    }
}

// ---------------- gcn_norm helpers ----------------
__global__ __launch_bounds__(256) void deg_init_kernel(float* deg, int n) {
    int i = blockIdx.x * blockDim.x + threadIdx.x;
    if (i < n) deg[i] = 1.0f;  // self loop
}

__global__ __launch_bounds__(256) void deg_count_kernel(const int* __restrict__ ei,
                                                        float* deg, int E) {
    int e = blockIdx.x * blockDim.x + threadIdx.x;
    if (e < E) unsafeAtomicAdd(&deg[ei[E + e]], 1.0f);  // dst row
}

__global__ __launch_bounds__(256) void deg_finish_kernel(float* deg, int n) {
    int i = blockIdx.x * blockDim.x + threadIdx.x;
    if (i < n) {
        float d = deg[i];
        deg[i] = (d > 0.0f) ? rsqrtf(d) : 0.0f;  // deg array becomes dinv
    }
}

__global__ __launch_bounds__(256) void edge_prep_kernel(const int* __restrict__ ei,
                                                        const float* __restrict__ dinv,
                                                        int* __restrict__ s32,
                                                        int* __restrict__ d32,
                                                        float* __restrict__ nrm, int E) {
    int e = blockIdx.x * blockDim.x + threadIdx.x;
    if (e < E) {
        int s = ei[e], d = ei[E + e];
        s32[e] = s;
        d32[e] = d;
        nrm[e] = dinv[s] * dinv[d];
    }
}

// ---------------- APPNP step ----------------
// hnext = ALPHA*h0 + (1-ALPHA)*dinv[i]^2*hcur   (restart + self-loop message, fully init)
__global__ __launch_bounds__(256) void step_init_kernel(const float4* __restrict__ h0,
                                                        const float4* __restrict__ hcur,
                                                        const float* __restrict__ dinv,
                                                        float4* __restrict__ hnext, int n) {
    int idx = blockIdx.x * blockDim.x + threadIdx.x;  // n * 32 float4 slots
    if (idx >= n * 32) return;
    int node = idx >> 5;
    float di = dinv[node];
    float sw = (1.0f - ALPHA) * di * di;
    float4 a = h0[idx], c = hcur[idx], o;
    o.x = ALPHA * a.x + sw * c.x;
    o.y = ALPHA * a.y + sw * c.y;
    o.z = ALPHA * a.z + sw * c.z;
    o.w = ALPHA * a.w + sw * c.w;
    hnext[idx] = o;
}

// One wave per edge: lane l handles features [4l, 4l+4). Gathers hit L2 (h fits in 192MB).
__global__ __launch_bounds__(256) void step_scatter_kernel(const int* __restrict__ s32,
                                                           const int* __restrict__ d32,
                                                           const float* __restrict__ nrm,
                                                           const float* __restrict__ hcur,
                                                           float* __restrict__ hnext, int E) {
    int gid  = blockIdx.x * blockDim.x + threadIdx.x;
    int edge = gid >> 5;
    int lane = gid & 31;
    if (edge >= E) return;
    int   s = s32[edge];
    int   d = d32[edge];
    float w = (1.0f - ALPHA) * nrm[edge];
    float4 v = ((const float4*)(hcur + (size_t)s * DIMS))[lane];
    float* hd = hnext + (size_t)d * DIMS + lane * 4;
    unsafeAtomicAdd(hd + 0, w * v.x);
    unsafeAtomicAdd(hd + 1, w * v.y);
    unsafeAtomicAdd(hd + 2, w * v.z);
    unsafeAtomicAdd(hd + 3, w * v.w);
}

// ---------------- launch ----------------
extern "C" void kernel_launch(void* const* d_in, const int* in_sizes, int n_in,
                              void* d_out, int out_size, void* d_ws, size_t ws_size,
                              hipStream_t stream) {
    const float* x  = (const float*)d_in[0];
    const int*   ei = (const int*)d_in[1];     // (2, E): [0:E)=src, [E:2E)=dst
    const float* W1 = (const float*)d_in[2];
    const float* b1 = (const float*)d_in[3];
    const float* W2 = (const float*)d_in[4];
    const float* b2 = (const float*)d_in[5];
    const int N = in_sizes[0] / DIMS;   // 100000
    const int E = in_sizes[1] / 2;      // 1600000
    float* out = (float*)d_out;

    char* ws = (char*)d_ws;
    size_t off = 0;
    auto alloc = [&](size_t bytes) {
        void* p = ws + off;
        off = (off + bytes + 511) & ~(size_t)511;
        return p;
    };
    float* dinv = (float*)alloc((size_t)N * 4);
    int*   s32  = (int*)  alloc((size_t)E * 4);
    int*   d32  = (int*)  alloc((size_t)E * 4);
    float* nrm  = (float*)alloc((size_t)E * 4);
    float* h0   = (float*)alloc((size_t)N * DIMS * 4);
    float* hA   = (float*)alloc((size_t)N * DIMS * 4);
    (void)ws_size; (void)n_in; (void)out_size;

    const int thr = 256;
    deg_init_kernel  <<<(N + thr - 1) / thr, thr, 0, stream>>>(dinv, N);
    deg_count_kernel <<<(E + thr - 1) / thr, thr, 0, stream>>>(ei, dinv, E);
    deg_finish_kernel<<<(N + thr - 1) / thr, thr, 0, stream>>>(dinv, N);
    edge_prep_kernel <<<(E + thr - 1) / thr, thr, 0, stream>>>(ei, dinv, s32, d32, nrm, E);

    const int tiles   = (N + 15) / 16;                      // 6250
    const int gblocks = (tiles + 8 * TPW - 1) / (8 * TPW);  // 98
    gemm_bias_kernel<true><<<gblocks, 256, 0, stream>>>(x, W1, b1, h0, N);

    float* bufs[2] = {hA, out};                // step k writes bufs[k&1]; step 9 -> out
    for (int k = 0; k < KSTEPS; ++k) {
        const float* hcur = (k == 0) ? h0 : bufs[(k - 1) & 1];
        float* hnext = bufs[k & 1];
        int slots = N * 32;                    // float4 slots
        step_init_kernel<<<(slots + thr - 1) / thr, thr, 0, stream>>>(
            (const float4*)h0, (const float4*)hcur, dinv, (float4*)hnext, N);
        long long sthreads = (long long)E * 32;
        step_scatter_kernel<<<(int)((sthreads + thr - 1) / thr), thr, 0, stream>>>(
            s32, d32, nrm, hcur, hnext, E);
    }
    // in-place: reads of each wave's 16x128 A block complete before its stores
    gemm_bias_kernel<false><<<gblocks, 256, 0, stream>>>(out, W2, b2, out, N);
}
